// AugmentedWindowAttention_15281493639412
// MI455X (gfx1250) — matrix operand order repaired and checked by reading, not compiled, not run on hardware
//
#include <hip/hip_runtime.h>
#include <hip/hip_bf16.h>

#define HH 12
#define EE 32
#define KK2 49
#define NWIN 64
#define BWIN 512
#define DIMM 384
#define MTOK (BWIN * KK2)          // 25088 token rows
#define THEADS (MTOK * HH)         // 301056 token-head rows
#define ATT_SCALE 0.17677669529663687f

typedef _Float16 v16h __attribute__((ext_vector_type(16)));
typedef _Float16 v8h  __attribute__((ext_vector_type(8)));
typedef float    v8f  __attribute__((ext_vector_type(8)));

__device__ __forceinline__ v16h cat8(v8h lo, v8h hi) {
  v16h r;
#pragma unroll
  for (int i = 0; i < 8; ++i) { r[i] = lo[i]; r[i + 8] = hi[i]; }
  return r;
}

__device__ __forceinline__ v8f wmma16(v16h a, v16h b, v8f c) {
  return __builtin_amdgcn_wmma_f32_16x16x32_f16(false, a, false, b, (short)0, c, false, false);
}

// ---------------- conversion / prep kernels ----------------

__global__ void cvt_f32_f16(const float* __restrict__ src, _Float16* __restrict__ dst, int n) {
  int i = blockIdx.x * blockDim.x + threadIdx.x;
  if (i < n) dst[i] = (_Float16)src[i];
}

// w_flat[f][e*32+c] = w_G[e*32+f][c]  (f<32, kk<1024), plus bias block:
// w_flat[f][1024+c] = w_biasgen[f][c]
__global__ void build_wflat(const float* __restrict__ wG, const float* __restrict__ wbg,
                            _Float16* __restrict__ wflat) {
  int i = blockIdx.x * blockDim.x + threadIdx.x;
  if (i >= 32 * 1056) return;
  int f = i / 1056, kk = i % 1056;
  float v;
  if (kk < 1024) {
    int e = kk >> 5, c = kk & 31;
    v = wG[(e * 32 + f) * 32 + c];
  } else {
    v = wbg[f * 32 + (kk - 1024)];
  }
  wflat[i] = (_Float16)v;
}

// comb[nw][h][q][k] = rpb_table[rel_idx[q*49+k]*12 + h] + mask[nw][q][k]
__global__ void build_bias(const float* __restrict__ tab, const int* __restrict__ rel,
                           const float* __restrict__ mask, float* __restrict__ out) {
  int i = blockIdx.x * blockDim.x + threadIdx.x;
  if (i >= NWIN * HH * KK2 * KK2) return;
  int r  = i % (KK2 * KK2);
  int nh = i / (KK2 * KK2);
  int h  = nh % HH;
  int nw = nh / HH;
  out[i] = tab[rel[r] * HH + h] + mask[nw * (KK2 * KK2) + r];
}

// ---------------- generic WMMA GEMM: C = act(A[M,K] @ W[N,K]^T + bias) ----------------
// block = 128 threads (4 waves); block tile 128(M) x 64(N); wave tile 32x64.
// Requires M%128==0, N%64==0, K%32==0.

template <bool OUT16, bool DO_GELU>
__global__ void gemm_wmma(const _Float16* __restrict__ A, const _Float16* __restrict__ W,
                          const float* __restrict__ bias, void* __restrict__ Cout,
                          int M, int N, int K) {
  const int wave = threadIdx.x >> 5;
  const int lane = threadIdx.x & 31;
  const int lm = lane & 15, lh = lane >> 4;
  const int nBlocks = N >> 6;
  const int bm = blockIdx.x / nBlocks, bn = blockIdx.x % nBlocks;
  const int m0 = bm * 128 + wave * 32;
  const int n0 = bn * 64;

  v8f acc[2][4] = {};
  const _Float16* arow0 = A + (size_t)(m0 + lm) * K + 8 * lh;
  const _Float16* arow1 = arow0 + (size_t)16 * K;

  for (int k0 = 0; k0 < K; k0 += 32) {
    v16h a0 = cat8(*(const v8h*)(arow0 + k0), *(const v8h*)(arow0 + k0 + 16));
    v16h a1 = cat8(*(const v8h*)(arow1 + k0), *(const v8h*)(arow1 + k0 + 16));
#pragma unroll
    for (int nt = 0; nt < 4; ++nt) {
      const _Float16* wrow = W + (size_t)(n0 + nt * 16 + lm) * K + k0 + 8 * lh;
      v16h b = cat8(*(const v8h*)wrow, *(const v8h*)(wrow + 16));
      acc[0][nt] = wmma16(a0, b, acc[0][nt]);
      acc[1][nt] = wmma16(a1, b, acc[1][nt]);
    }
  }
#pragma unroll
  for (int mt = 0; mt < 2; ++mt)
#pragma unroll
    for (int nt = 0; nt < 4; ++nt) {
      int n = n0 + nt * 16 + lm;
      float bv = bias[n];
#pragma unroll
      for (int j = 0; j < 8; ++j) {
        int m = m0 + mt * 16 + j + 8 * lh;
        float v = acc[mt][nt][j] + bv;
        if (DO_GELU) v = 0.5f * v * (1.0f + erff(v * 0.70710678118654752f));
        if (OUT16) ((_Float16*)Cout)[(size_t)m * N + n] = (_Float16)v;
        else       ((float*)Cout)[(size_t)m * N + n] = v;
      }
    }
}

// ---------------- windowed attention: one block per (window, head) ----------------
// qkv rows: [m][part*384 + h*32 + e], m = bw*49 + token. 4 waves x 16 query rows.

__global__ void attn_kernel(const _Float16* __restrict__ qkv, const float* __restrict__ comb,
                            _Float16* __restrict__ sa) {
  const int bw = blockIdx.x;
  const int h  = blockIdx.y;
  const int nw = bw & (NWIN - 1);
  const int tid = threadIdx.x;

  __shared__ _Float16 Vt[32 * 64];        // Vt[e][key], keys padded to 64
  __shared__ _Float16 P[4 * 16 * 64];     // per-wave softmax rows

  for (int idx = tid; idx < 32 * 64; idx += 128) {
    int e = idx >> 6, key = idx & 63;
    _Float16 v = (_Float16)0.0f;
    if (key < KK2) v = qkv[(size_t)(bw * KK2 + key) * 1152 + 768 + h * 32 + e];
    Vt[idx] = v;
  }
  __syncthreads();

  const int wave = tid >> 5, lane = tid & 31, lm = lane & 15, lh = lane >> 4;
  const int q0 = wave * 16;
  const float* bias = comb + (size_t)(nw * HH + h) * (KK2 * KK2);

  // Q fragment (rows clamped into valid range; padded rows never stored)
  int qm = q0 + lm; if (qm > KK2 - 1) qm = KK2 - 1;
  const _Float16* qrow = qkv + (size_t)(bw * KK2 + qm) * 1152 + h * 32 + 8 * lh;
  v16h aQ = cat8(*(const v8h*)qrow, *(const v8h*)(qrow + 16));

  v8f s[4];
#pragma unroll
  for (int nt = 0; nt < 4; ++nt) {
    int key = nt * 16 + lm; if (key > KK2 - 1) key = KK2 - 1;
    const _Float16* krow = qkv + (size_t)(bw * KK2 + key) * 1152 + 384 + h * 32 + 8 * lh;
    v16h bK = cat8(*(const v8h*)krow, *(const v8h*)(krow + 16));
    v8f z = {};
    s[nt] = wmma16(aQ, bK, z);
  }

  // scale + combined (rel-pos + mask) bias, pad columns -> -inf
  float p[4][8];
#pragma unroll
  for (int nt = 0; nt < 4; ++nt) {
    int kc = nt * 16 + lm;
#pragma unroll
    for (int j = 0; j < 8; ++j) {
      int q = q0 + j + 8 * lh; int qc = (q > KK2 - 1) ? (KK2 - 1) : q;
      float v;
      if (kc < KK2) v = s[nt][j] * ATT_SCALE + bias[qc * KK2 + kc];
      else          v = -1e30f;
      p[nt][j] = v;
    }
  }

  // softmax per row (row = fixed j within each 16-lane half)
#pragma unroll
  for (int j = 0; j < 8; ++j) {
    float mx = fmaxf(fmaxf(p[0][j], p[1][j]), fmaxf(p[2][j], p[3][j]));
#pragma unroll
    for (int off = 1; off < 16; off <<= 1) mx = fmaxf(mx, __shfl_xor(mx, off, 32));
    float sum = 0.0f;
#pragma unroll
    for (int nt = 0; nt < 4; ++nt) { p[nt][j] = __expf(p[nt][j] - mx); sum += p[nt][j]; }
#pragma unroll
    for (int off = 1; off < 16; off <<= 1) sum += __shfl_xor(sum, off, 32);
    float inv = 1.0f / sum;
#pragma unroll
    for (int nt = 0; nt < 4; ++nt) p[nt][j] *= inv;
  }

  // stage P (f16) for the P@V WMMA
  _Float16* Pw = &P[wave * 16 * 64];
#pragma unroll
  for (int nt = 0; nt < 4; ++nt)
#pragma unroll
    for (int j = 0; j < 8; ++j)
      Pw[(j + 8 * lh) * 64 + nt * 16 + lm] = (_Float16)p[nt][j];
  __syncthreads();

  v8f o[2] = {};
#pragma unroll
  for (int ks = 0; ks < 2; ++ks) {
    const _Float16* prow = Pw + lm * 64 + ks * 32 + 8 * lh;
    v16h aP = cat8(*(const v8h*)prow, *(const v8h*)(prow + 16));
#pragma unroll
    for (int nt = 0; nt < 2; ++nt) {
      const _Float16* vrow = Vt + (nt * 16 + lm) * 64 + ks * 32 + 8 * lh;
      v16h bV = cat8(*(const v8h*)vrow, *(const v8h*)(vrow + 16));
      o[nt] = wmma16(aP, bV, o[nt]);
    }
  }

#pragma unroll
  for (int nt = 0; nt < 2; ++nt)
#pragma unroll
    for (int j = 0; j < 8; ++j) {
      int q = q0 + j + 8 * lh;
      if (q < KK2)
        sa[(size_t)(bw * KK2 + q) * DIMM + h * 32 + nt * 16 + lm] = (_Float16)o[nt][j];
    }
}

// ---------------- fused hypernetwork reverse branch ----------------
// rev[t][f] = sum_{kk<1024} lg[t][kk/32]*gg[t][kk%32] * wflat[f][kk]
//           + sum_c gg[t][c] * wflat[f][1024+c]       (bias-gen, k-step 32)
// y[t*32+f] = sa[t*32+f] + rev[t][f]
// gg/lg/sa flattened as [THEADS][32]. 4 waves x 32 token-head rows (2 row groups).

__global__ void rev_kernel(const _Float16* __restrict__ gg, const _Float16* __restrict__ lg,
                           const _Float16* __restrict__ wflat, const _Float16* __restrict__ sa,
                           _Float16* __restrict__ y) {
  const int wave = threadIdx.x >> 5, lane = threadIdx.x & 31;
  const int lm = lane & 15, lh = lane >> 4;
  const size_t t0 = (size_t)blockIdx.x * 128 + wave * 32;
  const size_t ta = t0 + lm;        // row group 0
  const size_t tb = ta + 16;        // row group 1

  const _Float16* ga = gg + 32 * ta + 8 * lh;
  const _Float16* gb = gg + 32 * tb + 8 * lh;
  v16h agg0 = cat8(*(const v8h*)ga, *(const v8h*)(ga + 16));
  v16h agg1 = cat8(*(const v8h*)gb, *(const v8h*)(gb + 16));
  const v16h* la = (const v16h*)(lg + 32 * ta);
  const v16h* lb = (const v16h*)(lg + 32 * tb);
  v16h lg0a = la[0], lg1a = la[1];
  v16h lg0b = lb[0], lg1b = lb[1];

  v8f acc[2][2] = {};
#pragma unroll
  for (int ks = 0; ks < 33; ++ks) {
    v16h a0, a1;
    if (ks < 32) {
      _Float16 lva = (ks < 16) ? lg0a[ks] : lg1a[ks - 16];
      _Float16 lvb = (ks < 16) ? lg0b[ks] : lg1b[ks - 16];
#pragma unroll
      for (int i = 0; i < 16; ++i) { a0[i] = agg0[i] * lva; a1[i] = agg1[i] * lvb; }
    } else {
      a0 = agg0; a1 = agg1;                                // bias-gen block, lg == 1
    }
    const int kk0 = (ks < 32) ? ks * 32 : 1024;
#pragma unroll
    for (int nt = 0; nt < 2; ++nt) {
      const _Float16* brow = wflat + (size_t)(nt * 16 + lm) * 1056 + kk0 + 8 * lh;
      v16h b = cat8(*(const v8h*)brow, *(const v8h*)(brow + 16));
      acc[0][nt] = wmma16(a0, b, acc[0][nt]);
      acc[1][nt] = wmma16(a1, b, acc[1][nt]);
    }
  }

#pragma unroll
  for (int mt = 0; mt < 2; ++mt)
#pragma unroll
    for (int nt = 0; nt < 2; ++nt)
#pragma unroll
      for (int j = 0; j < 8; ++j) {
        size_t t = t0 + mt * 16 + j + 8 * lh;
        int f = nt * 16 + lm;
        float v = (float)sa[32 * t + f] + acc[mt][nt][j];
        y[32 * t + f] = (_Float16)v;
      }
}

// ---------------- host side ----------------

extern "C" void kernel_launch(void* const* d_in, const int* in_sizes, int n_in,
                              void* d_out, int out_size, void* d_ws, size_t ws_size,
                              hipStream_t stream) {
  const float* x        = (const float*)d_in[0];
  const float* mask     = (const float*)d_in[1];
  const float* w_qkv    = (const float*)d_in[2];
  const float* b_qkv    = (const float*)d_in[3];
  const float* rpb_tab  = (const float*)d_in[4];
  const float* w_global = (const float*)d_in[5];
  const float* b_global = (const float*)d_in[6];
  const float* w_local  = (const float*)d_in[7];
  const float* b_local  = (const float*)d_in[8];
  const float* w_G      = (const float*)d_in[9];
  const float* w_bg     = (const float*)d_in[10];
  const float* w_proj   = (const float*)d_in[11];
  const float* b_proj   = (const float*)d_in[12];
  const int*   rel_idx  = (const int*)d_in[13];
  float* out = (float*)d_out;

  char* ws = (char*)d_ws;
  auto alloc = [&](size_t bytes) -> char* {
    char* p = ws;
    ws += (bytes + 255) & ~(size_t)255;
    return p;
  };

  _Float16* xh   = (_Float16*)alloc((size_t)MTOK * DIMM * 2);
  _Float16* wqh  = (_Float16*)alloc((size_t)1152 * 384 * 2);
  _Float16* wgh  = (_Float16*)alloc((size_t)384 * 384 * 2);
  _Float16* wlh  = (_Float16*)alloc((size_t)384 * 384 * 2);
  _Float16* wph  = (_Float16*)alloc((size_t)384 * 384 * 2);
  _Float16* wfl  = (_Float16*)alloc((size_t)32 * 1056 * 2);
  float*    comb = (float*)alloc((size_t)NWIN * HH * KK2 * KK2 * 4);
  _Float16* sah  = (_Float16*)alloc((size_t)MTOK * DIMM * 2);
  _Float16* qkvh = (_Float16*)alloc((size_t)MTOK * 1152 * 2);
  // qkvh region is dead after attention: reuse it for gg / lg / y (3 x MTOK*384 == MTOK*1152)
  _Float16* ggh = qkvh;
  _Float16* lgh = qkvh + (size_t)MTOK * DIMM;
  _Float16* yh  = qkvh + (size_t)2 * MTOK * DIMM;

  // --- conversions & precomputation ---
  {
    int n = MTOK * DIMM;
    cvt_f32_f16<<<(n + 255) / 256, 256, 0, stream>>>(x, xh, n);
  }
  cvt_f32_f16<<<(1152 * 384 + 255) / 256, 256, 0, stream>>>(w_qkv, wqh, 1152 * 384);
  cvt_f32_f16<<<(384 * 384 + 255) / 256, 256, 0, stream>>>(w_global, wgh, 384 * 384);
  cvt_f32_f16<<<(384 * 384 + 255) / 256, 256, 0, stream>>>(w_local, wlh, 384 * 384);
  cvt_f32_f16<<<(384 * 384 + 255) / 256, 256, 0, stream>>>(w_proj, wph, 384 * 384);
  build_wflat<<<(32 * 1056 + 255) / 256, 256, 0, stream>>>(w_G, w_bg, wfl);
  {
    int n = NWIN * HH * KK2 * KK2;
    build_bias<<<(n + 255) / 256, 256, 0, stream>>>(rpb_tab, rel_idx, mask, comb);
  }

  // --- QKV projection: [25088,384] @ [1152,384]^T -> f16 [25088,1152] ---
  gemm_wmma<true, false><<<(MTOK / 128) * (1152 / 64), 128, 0, stream>>>(
      xh, wqh, b_qkv, qkvh, MTOK, 1152, 384);

  // --- windowed attention -> sa f16 [25088,384] ---
  {
    dim3 grid(BWIN, HH);
    attn_kernel<<<grid, 128, 0, stream>>>(qkvh, comb, sah);
  }

  // --- global_groups = gelu(sa @ w_global^T + b) -> ggh (overwrites qkv region) ---
  gemm_wmma<true, true><<<(MTOK / 128) * (DIMM / 64), 128, 0, stream>>>(
      sah, wgh, b_global, ggh, MTOK, DIMM, 384);

  // --- local_groups = gelu(x @ w_local^T + b) -> lgh ---
  gemm_wmma<true, true><<<(MTOK / 128) * (DIMM / 64), 128, 0, stream>>>(
      xh, wlh, b_local, lgh, MTOK, DIMM, 384);

  // --- fused hypernetwork reverse branch: y = sa + rev ---
  rev_kernel<<<THEADS / 128, 128, 0, stream>>>(ggh, lgh, wfl, sah, yh);

  // --- output projection: out f32 = y @ w_proj^T + b_proj ---
  gemm_wmma<false, false><<<(MTOK / 128) * (DIMM / 64), 128, 0, stream>>>(
      yh, wph, b_proj, out, MTOK, DIMM, 384);
}
